// VQModel_84456236909037
// MI455X (gfx1250) — matrix operand order repaired and checked by reading, not compiled, not run on hardware
//
#include <hip/hip_runtime.h>

// ---------------------------------------------------------------------------
// VQModel forward for MI455X (gfx1250, wave32, WMMA).
//
// Sizes (baked from reference): B=16, C=128, H=W=32 -> NTOK=16384, E=64,
// NCODE=8192, delta=10, ALPHA=0.5, BETA_FRAC=0.5, VQ_BETA=0.25.
//
// Hot spot: token-vs-codebook distance GEMM, M=16384 N=8192 K=64 (17.2 GFLOP)
// -> v_wmma_f32_16x16x32_f16 (f16 operands, f32 accumulate), top-10 selection
// fused (512MB distance matrix never materialized).
//   * wave-uniform threshold prefilter: skip LDS spill + top-10 scan for
//     chunks with no possible top-10 survivor (most chunks after warmup).
//   * async Global->LDS double-buffered codebook staging (ASYNCcnt) via
//     __builtin_amdgcn_global_load_async_to_lds_b128 (typed global-int4* /
//     local-int4* per the clang-22 signature).
// ---------------------------------------------------------------------------

typedef __attribute__((ext_vector_type(16))) _Float16 v16h;
typedef __attribute__((ext_vector_type(8)))  float    v8f;

#define NB     16
#define NC     128
#define HW     1024
#define NE     64
#define NCODE  8192
#define NTOK   (NB * HW)          // 16384
#define DEC_ELTS (NB * NC * HW)   // 2097152

// ---- workspace layout (bytes, all 256-aligned) ----
#define OFF_ZF    0                          // f32 [NTOK][64]   4 MB
#define OFF_ZFH   (OFF_ZF  + NTOK*64*4)      // f16 [NTOK][64]   2 MB
#define OFF_CBH   (OFF_ZFH + NTOK*64*2)      // f16 [NCODE][64]  1 MB
#define OFF_CN    (OFF_CBH + NCODE*64*2)     // f32 [NCODE]      32 KB
#define OFF_SEL   (OFF_CN  + NCODE*4)        // i32 [NTOK]       64 KB
#define OFF_QP    (OFF_SEL + NTOK*4)         // f32 [NTOK]       64 KB

// ---- async Global->LDS availability probe (compile-time) ----
#if defined(__HIP_DEVICE_COMPILE__) && __has_builtin(__builtin_amdgcn_global_load_async_to_lds_b128)
#define ASYNC_STAGE 1
#else
#define ASYNC_STAGE 0
#endif

// ---------------------------------------------------------------------------
// Kernel 1: codebook squared norms + f16 copy.
// ---------------------------------------------------------------------------
__global__ void prep_codes(const float* __restrict__ cbf,
                           _Float16* __restrict__ cb_h,
                           float* __restrict__ cn) {
  int n = blockIdx.x * 256 + threadIdx.x;           // 32 blocks x 256
  const float* row = cbf + (size_t)n * 64;
  float s = 0.f;
#pragma unroll
  for (int e = 0; e < 64; ++e) {
    float v = row[e];
    s += v * v;
    cb_h[(size_t)n * 64 + e] = (_Float16)v;
  }
  cn[n] = s;
}

// ---------------------------------------------------------------------------
// Kernel 2: quant_conv (1x1): zf[tok][e] = qb[e] + sum_c h[b][c][hw]*qw[e][c]
// WG = 256 threads handles 32 consecutive tokens (same batch image).
// ---------------------------------------------------------------------------
__global__ void prep_zf(const float* __restrict__ h,
                        const float* __restrict__ qw,   // [64][128]
                        const float* __restrict__ qb,   // [64]
                        float* __restrict__ zf,
                        _Float16* __restrict__ zf_h) {
  __shared__ __align__(16) float s_h[128][36];          // pad 36: 144B rows
  const int t = threadIdx.x;
  const int tokb = blockIdx.x * 32;                     // 512 blocks
  const int b   = tokb >> 10;
  const int hwb = tokb & 1023;

  {   // stage h: thread t loads 16 floats of row c = t>>1, half hs = t&1
    int c = t >> 1, hs = t & 1;
    const float4* src = (const float4*)(h + ((size_t)(b * 128 + c)) * 1024 + hwb + hs * 16);
    float4* dst = (float4*)(&s_h[c][hs * 16]);
#pragma unroll
    for (int q = 0; q < 4; ++q) dst[q] = src[q];
  }
  __syncthreads();

  const int e    = t & 63;
  const int isub = t >> 6;                              // 0..3
  float acc[8];
  const float bias = qb[e];
#pragma unroll
  for (int r = 0; r < 8; ++r) acc[r] = bias;
  for (int c = 0; c < 128; ++c) {
    float w = qw[e * 128 + c];
#pragma unroll
    for (int r = 0; r < 8; ++r) acc[r] += s_h[c][r * 4 + isub] * w;  // LDS bcast
  }
#pragma unroll
  for (int r = 0; r < 8; ++r) {
    int tok = tokb + r * 4 + isub;
    zf[(size_t)tok * 64 + e]   = acc[r];
    zf_h[(size_t)tok * 64 + e] = (_Float16)acc[r];
  }
}

// ---------------------------------------------------------------------------
// Codebook chunk staging: 64 codes x 64 f16 into padded LDS rows (72 halves,
// 144B -> 16B-aligned segments, conflict-free B-fragment reads).
// Async path: each of 128 threads issues 4x GLOBAL_LOAD_ASYNC_TO_LDS_B128
// (512 x 16B = 8KB), tracked by ASYNCcnt. Fallback: uint4 copy via VGPRs.
// ---------------------------------------------------------------------------
#if ASYNC_STAGE
typedef int async_b128 __attribute__((vector_size(16)));        // int4
typedef __attribute__((address_space(1))) async_b128 gbl_b128;  // global
typedef __attribute__((address_space(3))) async_b128 lds_b128;  // LDS
#endif

__device__ __forceinline__ void stage_cb_chunk(const _Float16* __restrict__ cb_h,
                                               int cb0,
                                               _Float16 (&buf)[64][72],
                                               int t) {
#if ASYNC_STAGE
#pragma unroll
  for (int qtr = 0; qtr < 4; ++qtr) {
    int idx = qtr * 128 + t;                 // 0..511 16B segments
    int row = idx >> 3, seg = idx & 7;
    gbl_b128* g = (gbl_b128*)(cb_h + (size_t)(cb0 + row) * 64 + (seg << 3));
    lds_b128* l = (lds_b128*)(&buf[row][seg << 3]);
    __builtin_amdgcn_global_load_async_to_lds_b128(g, l, 0, 0);
  }
#else
  int row = t >> 1, hs = t & 1;
  const uint4* src = (const uint4*)(cb_h + (size_t)(cb0 + row) * 64 + hs * 32);
  uint4* dst = (uint4*)(&buf[row][hs * 32]);
  dst[0] = src[0]; dst[1] = src[1]; dst[2] = src[2]; dst[3] = src[3];
#endif
}

__device__ __forceinline__ void wait_async_stage() {
#if ASYNC_STAGE
#if __has_builtin(__builtin_amdgcn_s_wait_asynccnt)
  __builtin_amdgcn_s_wait_asynccnt(0);
#else
  asm volatile("s_wait_asynccnt 0" ::: "memory");
#endif
#endif
}

// ---------------------------------------------------------------------------
// Kernel 3 (HOT): fused distance GEMM + per-token top-10 + selection + qloss.
// Block = 128 threads (4 waves); wave owns 16 tokens, WG owns 64.
// score s[tok][n] = ||cb_n||^2 - 2*<zf_tok, cb_n>  (||zf||^2 row-constant).
// ---------------------------------------------------------------------------
#define CSWAP(da, db, ia, ib) \
  do { if ((db) < (da)) { float _tf = (da); (da) = (db); (db) = _tf; \
                          int _ti = (ia); (ia) = (ib); (ib) = _ti; } } while (0)
#define TOP10_INSERT(v, g)                                        \
  do { if ((v) < d9) {                                            \
        d9 = (v); j9 = (g);                                       \
        CSWAP(d8, d9, j8, j9); CSWAP(d7, d8, j7, j8);             \
        CSWAP(d6, d7, j6, j7); CSWAP(d5, d6, j5, j6);             \
        CSWAP(d4, d5, j4, j5); CSWAP(d3, d4, j3, j4);             \
        CSWAP(d2, d3, j2, j3); CSWAP(d1, d2, j1, j2);             \
        CSWAP(d0, d1, j0, j1); } } while (0)

__global__ void vq_topk(const _Float16* __restrict__ zf_h,
                        const _Float16* __restrict__ cb_h,
                        const float* __restrict__ cn,
                        const float* __restrict__ zf,
                        const float* __restrict__ cbf,
                        const float* __restrict__ rprob,
                        const int*   __restrict__ ridxin,
                        int*   __restrict__ sel_out,
                        float* __restrict__ qpart) {
  __shared__ __align__(16) _Float16 s_cb[2][64][72];  // double-buffered chunk
  __shared__ float s_scores[4][16][65];               // per-wave spill tiles
  __shared__ float s_md[4][16][10];
  __shared__ int   s_mi[4][16][10];

  const int t    = threadIdx.x;
  const int lane = t & 31;
  const int w    = t >> 5;                          // wave 0..3
  const int col  = lane & 15;
  const int hi   = lane >> 4;
  const int tokbase = blockIdx.x * 64 + w * 16;     // 256 blocks

  // ---- A fragments (16 tokens x 64 K, f16), resident for whole kernel.
  // A 16x32 layout (matches B): row = lane%16, halves q -> K = hi*16 + q
  // within each 32-K chunk.
  v16h a0, a1;
  {
    const _Float16* za = zf_h + (size_t)(tokbase + col) * 64;
#pragma unroll
    for (int q = 0; q < 16; ++q) {
      int k = hi * 16 + q;
      a0[q] = za[k];
      a1[q] = za[32 + k];
    }
  }

  // per-lane sorted top-10 (ascending); lanes 0-15 cover cols 0..31 of their
  // token row per chunk, lanes 16-31 cover cols 32..63; merged at the end.
  float d0 = 3.0e38f, d1 = 3.0e38f, d2 = 3.0e38f, d3 = 3.0e38f, d4 = 3.0e38f,
        d5 = 3.0e38f, d6 = 3.0e38f, d7 = 3.0e38f, d8 = 3.0e38f, d9 = 3.0e38f;
  int j0 = 0, j1 = 0, j2 = 0, j3 = 0, j4 = 0, j5 = 0, j6 = 0, j7 = 0, j8 = 0, j9 = 0;
  float thr = 3.0e38f;       // wave-uniform: max over lanes of current d9

  // prologue: stage chunk 0
  stage_cb_chunk(cb_h, 0, s_cb[0], t);
  wait_async_stage();
  __syncthreads();

  int p = 0;
  for (int ci = 0; ci < NCODE / 64; ++ci) {
    const int cb0 = ci * 64;
    if (ci + 1 < NCODE / 64)                         // overlap next-chunk DMA
      stage_cb_chunk(cb_h, cb0 + 64, s_cb[p ^ 1], t);

    // ---- 4 subtiles of 16 codes, K=64 -> 2 WMMA each; scores kept in VGPRs.
    float sv[4][8];
#pragma unroll
    for (int st = 0; st < 4; ++st) {
      // B 32x16 f16 layout: column = lane%16; halves j -> K = hi*16 + j
      // within each 32-K chunk.
      const _Float16* bp = &s_cb[p][st * 16 + col][hi * 16];
      v16h b0f, b1f;
#pragma unroll
      for (int q = 0; q < 16; ++q) { b0f[q] = bp[q]; b1f[q] = bp[32 + q]; }

      v8f acc = {};
      acc = __builtin_amdgcn_wmma_f32_16x16x32_f16(false, a0, false, b0f,
                                                   (short)0, acc, false, false);
      acc = __builtin_amdgcn_wmma_f32_16x16x32_f16(false, a1, false, b1f,
                                                   (short)0, acc, false, false);
      float cnv = cn[cb0 + st * 16 + col];           // L2-resident
#pragma unroll
      for (int q = 0; q < 8; ++q)                    // D row m = q + 8*hi
        sv[st][q] = __builtin_fmaf(-2.0f, acc[q], cnv);
    }

    // ---- threshold prefilter: can any score enter any lane's top-10?
    float lmin = sv[0][0];
#pragma unroll
    for (int st = 0; st < 4; ++st)
#pragma unroll
      for (int q = 0; q < 8; ++q) lmin = fminf(lmin, sv[st][q]);

    if (__any(lmin < thr)) {                         // wave-uniform branch
      // spill 16x64 score tile to LDS (padded rows -> conflict-free)
#pragma unroll
      for (int st = 0; st < 4; ++st)
#pragma unroll
        for (int q = 0; q < 8; ++q)
          s_scores[w][q + 8 * hi][st * 16 + col] = sv[st][q];

      // scan my token row, my half of the 64 columns
      const int m = lane & 15;
      const int nb = hi * 32;
#pragma unroll 4
      for (int q = 0; q < 32; ++q) {
        float v = s_scores[w][m][nb + q];
        int g = cb0 + nb + q;
        TOP10_INSERT(v, g);
      }
      // refresh wave-uniform threshold = max over lanes of d9
      float mx = d9;
#pragma unroll
      for (int off = 16; off > 0; off >>= 1)
        mx = fmaxf(mx, __shfl_xor(mx, off, 32));
      thr = mx;
    }

    wait_async_stage();        // my next-chunk DMA segments landed
    __syncthreads();           // everyone done reading buf[p] / writing buf[p^1]
    p ^= 1;
  }

  // ---- merge upper-half lists into lower-half lanes.
  const int m = lane & 15;
  if (hi == 1) {
    s_md[w][m][0] = d0; s_md[w][m][1] = d1; s_md[w][m][2] = d2; s_md[w][m][3] = d3;
    s_md[w][m][4] = d4; s_md[w][m][5] = d5; s_md[w][m][6] = d6; s_md[w][m][7] = d7;
    s_md[w][m][8] = d8; s_md[w][m][9] = d9;
    s_mi[w][m][0] = j0; s_mi[w][m][1] = j1; s_mi[w][m][2] = j2; s_mi[w][m][3] = j3;
    s_mi[w][m][4] = j4; s_mi[w][m][5] = j5; s_mi[w][m][6] = j6; s_mi[w][m][7] = j7;
    s_mi[w][m][8] = j8; s_mi[w][m][9] = j9;
  }
  __syncthreads();
  if (hi == 0) {
#pragma unroll
    for (int k = 0; k < 10; ++k) {
      float v = s_md[w][m][k];
      int g = s_mi[w][m][k];
      TOP10_INSERT(v, g);
    }
    const int tok = tokbase + m;
    const int b = tok >> 10;
    // ridx = (batch < 8) ? (rand_prob > 0.5 ? 0 : rand_idx) : 0 (argmin)
    int r = 0;
    if (b < 8) {
      float rp = rprob[tok];
      int ri = ridxin[tok];
      r = (rp > 0.5f) ? 0 : ri;
      r = (r > 9) ? 9 : r;
    }
    int selv = j0;
    selv = (r == 1) ? j1 : selv; selv = (r == 2) ? j2 : selv;
    selv = (r == 3) ? j3 : selv; selv = (r == 4) ? j4 : selv;
    selv = (r == 5) ? j5 : selv; selv = (r == 6) ? j6 : selv;
    selv = (r == 7) ? j7 : selv; selv = (r == 8) ? j8 : selv;
    selv = (r == 9) ? j9 : selv;
    sel_out[tok] = selv;

    // exact f32 commitment term for the argmin code (matches reference mean)
    const float* zrow = zf  + (size_t)tok * 64;
    const float* crow = cbf + (size_t)j0 * 64;
    float s = 0.f;
#pragma unroll
    for (int e = 0; e < 64; ++e) {
      float dlt = crow[e] - zrow[e];
      s += dlt * dlt;
    }
    qpart[tok] = s;
  }
}

// ---------------------------------------------------------------------------
// Kernel 4: post_quant_conv: dec[b][c][hw] = pb[c] + sum_e cb[sel][e]*pw[c][e]
// ---------------------------------------------------------------------------
__global__ void dec_gemm(const int* __restrict__ sel,
                         const float* __restrict__ cbf,
                         const float* __restrict__ pw,   // [128][64]
                         const float* __restrict__ pb,   // [128]
                         float* __restrict__ dec) {
  __shared__ float sQ[32][65];
  const int t = threadIdx.x;
  const int tokb = blockIdx.x * 32;                      // 512 blocks
  {
    int i = t >> 3, part = t & 7;
    int sv = sel[tokb + i];
    const float* src = cbf + (size_t)sv * 64 + part * 8;
#pragma unroll
    for (int q = 0; q < 8; ++q) sQ[i][part * 8 + q] = src[q];
  }
  __syncthreads();
  const int lane = t & 31;
  const int tok = tokb + lane;
  const int b = tok >> 10;
  const int hw = tok & 1023;
  for (int c = t >> 5; c < 128; c += 8) {                // c uniform per wave
    float acc = pb[c];
    const float* wr = pw + c * 64;                       // scalar-uniform row
#pragma unroll
    for (int e = 0; e < 64; ++e) acc += sQ[lane][e] * wr[e];
    dec[(size_t)b * (NC * HW) + (size_t)c * HW + hw] = acc;
  }
}

// ---------------------------------------------------------------------------
// Kernel 5: deterministic qloss reduction.
// qloss = (1 + VQ_BETA) * sum(qpart) / (NTOK * 64)
// ---------------------------------------------------------------------------
__global__ void qloss_reduce(const float* __restrict__ qpart,
                             float* __restrict__ out) {
  __shared__ float red[256];
  const int t = threadIdx.x;
  float s = 0.f;
  for (int i = t; i < NTOK; i += 256) s += qpart[i];
  red[t] = s;
  __syncthreads();
  for (int off = 128; off > 0; off >>= 1) {
    if (t < off) red[t] += red[t + off];
    __syncthreads();
  }
  if (t == 0) out[0] = 1.25f * red[0] / ((float)NTOK * 64.0f);
}

// ---------------------------------------------------------------------------
extern "C" void kernel_launch(void* const* d_in, const int* in_sizes, int n_in,
                              void* d_out, int out_size, void* d_ws, size_t ws_size,
                              hipStream_t stream) {
  const float* h     = (const float*)d_in[0];
  const float* qw    = (const float*)d_in[1];
  const float* qb    = (const float*)d_in[2];
  const float* cbf   = (const float*)d_in[3];
  const float* pw    = (const float*)d_in[4];
  const float* pb    = (const float*)d_in[5];
  const float* rprob = (const float*)d_in[6];
  const int*   ridx  = (const int*)d_in[7];
  // d_in[8] = delta (==10, baked into TOP10 machinery)

  char* ws = (char*)d_ws;
  float*    zf   = (float*)   (ws + OFF_ZF);
  _Float16* zf_h = (_Float16*)(ws + OFF_ZFH);
  _Float16* cb_h = (_Float16*)(ws + OFF_CBH);
  float*    cn   = (float*)   (ws + OFF_CN);
  int*      sel  = (int*)     (ws + OFF_SEL);
  float*    qp   = (float*)   (ws + OFF_QP);
  float*    out  = (float*)d_out;

  prep_codes  <<<NCODE / 256, 256, 0, stream>>>(cbf, cb_h, cn);
  prep_zf     <<<NTOK / 32,   256, 0, stream>>>(h, qw, qb, zf, zf_h);
  vq_topk     <<<NTOK / 64,   128, 0, stream>>>(zf_h, cb_h, cn, zf, cbf,
                                                rprob, ridx, sel, qp);
  dec_gemm    <<<NTOK / 32,   256, 0, stream>>>(sel, cbf, pw, pb, out);
  qloss_reduce<<<1,           256, 0, stream>>>(qp, out + DEC_ELTS);
}
